// ThreadSafeIntelSpecificAttention_13932873908359
// MI455X (gfx1250) — matrix operand order repaired and checked by reading, compiled, not run on hardware
//
#include <hip/hip_runtime.h>
#include <hip/hip_bf16.h>

// ---------------------------------------------------------------------------
// GQA attention block for MI455X (gfx1250, wave32, WMMA, async-to-LDS).
//   B=2, S=2048, HIDDEN=2048, NH=16, NKV=4, HD=128, NREP=4
// Pipeline: f32->bf16 convert (LDS-tiled weight transpose) -> QKV GEMM
//           (WMMA bf16, double-buffered async LDS staging, top-of-loop wait)
//           -> RoPE -> flash attention (WMMA bf16, online softmax,
//           double-buffered async K/V staging) -> Wo GEMM (f32 out)
// ---------------------------------------------------------------------------

typedef __attribute__((ext_vector_type(16))) __bf16 v16bf;
typedef __attribute__((ext_vector_type(8)))  __bf16 bh8;
typedef __attribute__((ext_vector_type(8)))  float  v8f;

#define BATCH   2
#define SEQ     2048
#define HIDDEN  2048
#define NH      16
#define NKV     4
#define HD      128
#define MROWS   (BATCH*SEQ)   /* 4096 */
#define NQKV    3072          /* 2048 q | 512 k | 512 v */

// workspace layout (bytes)
constexpr size_t XB_OFF   = 0;
constexpr size_t XB_SZ    = (size_t)MROWS * HIDDEN * 2;        // X bf16 [m][k]
constexpr size_t WQKV_OFF = XB_OFF + XB_SZ;
constexpr size_t WQKV_SZ  = (size_t)NQKV * HIDDEN * 2;         // Wt bf16 [n][k]
constexpr size_t WO_OFF   = WQKV_OFF + WQKV_SZ;
constexpr size_t WO_SZ    = (size_t)HIDDEN * HIDDEN * 2;       // Wot bf16 [n][k]
constexpr size_t QB_OFF   = WO_OFF + WO_SZ;
constexpr size_t QB_SZ    = (size_t)BATCH * NH * SEQ * HD * 2; // Q [b,h,s,d]
constexpr size_t KB_OFF   = QB_OFF + QB_SZ;
constexpr size_t KB_SZ    = (size_t)BATCH * NKV * SEQ * HD * 2;// K [b,kvh,s,d]
constexpr size_t VB_OFF   = KB_OFF + KB_SZ;
constexpr size_t VB_SZ    = KB_SZ;                             // V [b,kvh,d,s]
constexpr size_t OB_OFF   = VB_OFF + VB_SZ;
constexpr size_t OB_SZ    = (size_t)MROWS * HIDDEN * 2;        // attn out bf16

__device__ inline v8f zero8() {
    v8f z;
#pragma unroll
    for (int i = 0; i < 8; ++i) z[i] = 0.0f;
    return z;
}

// ---- gfx1250 async global->LDS copy (16B per lane), ASYNCcnt-tracked ------
// Generic shared pointers carry the LDS byte address in their low 32 bits
// (ISA 10.2: LDS aperture truncates to addr[31:0]).
__device__ inline void async_copy16(const __bf16* gbase, unsigned voff_bytes,
                                    void* lds_ptr) {
    unsigned lds = (unsigned)(unsigned long long)lds_ptr;
    asm volatile("global_load_async_to_lds_b128 %0, %1, %2"
                 :: "v"(lds), "v"(voff_bytes), "s"(gbase)
                 : "memory");
}
__device__ inline void wait_async0() {
    asm volatile("s_wait_asynccnt 0x0" ::: "memory");
}

// A fragment (16x32 bf16): lane L -> row (L&15),
// 16 consecutive K at (L>>4)*16 (matches load_b_frag).
__device__ inline v16bf load_a_frag(const __bf16* p, int ld, int row0, int k0, int lane) {
    const __bf16* q = p + (size_t)(row0 + (lane & 15)) * ld + k0 + ((lane >> 4) << 4);
    bh8 lo = *(const bh8*)(q);
    bh8 hi = *(const bh8*)(q + 8);
    v16bf r;
#pragma unroll
    for (int i = 0; i < 8; ++i) { r[i] = lo[i]; r[i + 8] = hi[i]; }
    return r;
}

// B fragment (32x16 bf16): lane L -> col (L&15), 16 consecutive K at (L>>4)*16.
// `p` is stored N-major: p[n*ld + k] -> one contiguous 32B LDS read per lane.
__device__ inline v16bf load_b_frag(const __bf16* p, int ld, int n0, int k0, int lane) {
    return *(const v16bf*)(p + (size_t)(n0 + (lane & 15)) * ld + k0 + ((lane >> 4) << 4));
}

// ---------------------------------------------------------------------------
// 0a) X: coalesced fp32 -> bf16 (4 elems/thread)
// ---------------------------------------------------------------------------
__global__ void convert_x_kernel(const float* __restrict__ X, __bf16* __restrict__ Xb) {
    size_t i = ((size_t)blockIdx.x * blockDim.x + threadIdx.x) * 4;
    if (i >= (size_t)MROWS * HIDDEN) return;
    float4 v = *(const float4*)(X + i);
    Xb[i + 0] = (__bf16)v.x;
    Xb[i + 1] = (__bf16)v.y;
    Xb[i + 2] = (__bf16)v.z;
    Xb[i + 3] = (__bf16)v.w;
}

// ---------------------------------------------------------------------------
// 0b) LDS-tiled transpose+convert: dst[n][k] (bf16) from src[k][n] (f32).
//     64x64 tiles; coalesced float4 reads along n, coalesced 16B writes
//     along k. qkv_mode=1 selects Wq/Wk/Wv by n-range (boundaries 64-aligned).
// ---------------------------------------------------------------------------
__global__ __launch_bounds__(256) void transpose_convert_kernel(
    const float* __restrict__ Wq, const float* __restrict__ Wk,
    const float* __restrict__ Wv, __bf16* __restrict__ dst, int qkv_mode) {
    __shared__ __bf16 tile[64][72];                 // [n][k], +8 pad
    const int k0 = blockIdx.y * 64;
    const int n0 = blockIdx.x * 64;
    const int tid = threadIdx.x;

    const float* src; int ld, nbase;
    if (!qkv_mode)      { src = Wq; ld = HIDDEN; nbase = n0; }
    else if (n0 < 2048) { src = Wq; ld = 2048;   nbase = n0; }
    else if (n0 < 2560) { src = Wk; ld = 512;    nbase = n0 - 2048; }
    else                { src = Wv; ld = 512;    nbase = n0 - 2560; }

#pragma unroll
    for (int it = 0; it < 4; ++it) {                // read 64x64 f32
        int r = (tid >> 4) + it * 16;               // k within tile
        int c = (tid & 15) << 2;                    // n within tile
        float4 v = *(const float4*)(src + (size_t)(k0 + r) * ld + nbase + c);
        tile[c + 0][r] = (__bf16)v.x;
        tile[c + 1][r] = (__bf16)v.y;
        tile[c + 2][r] = (__bf16)v.z;
        tile[c + 3][r] = (__bf16)v.w;
    }
    __syncthreads();
#pragma unroll
    for (int it = 0; it < 2; ++it) {                // write 64 rows x 64 k
        int idx = tid + it * 256;                   // 512 chunks of 8
        int n = idx >> 3, kc = (idx & 7) << 3;
        bh8 o;
#pragma unroll
        for (int i = 0; i < 8; ++i) o[i] = tile[n][kc + i];
        *(bh8*)(dst + (size_t)(n0 + n) * HIDDEN + k0 + kc) = o;
    }
}

// ---------------------------------------------------------------------------
// shared GEMM mainloop: C[128x128] tile = A[m][k] x Wt[n][k]^T, bf16 WMMA.
// Double-buffered async staging, wait at TOP of loop so tile t+1's loads
// overlap the whole compute of tile t. 8 waves, wave tile 64x32.
// ---------------------------------------------------------------------------
struct GemmAcc { v8f a[4][2]; };

__device__ inline void gemm_mainloop(const __bf16* __restrict__ A,
                                     const __bf16* __restrict__ Wt,
                                     int m0, int n0, int tid, int lane,
                                     int wm, int wn,
                                     __bf16 (*sA)[128 * 32], __bf16 (*sB)[128 * 32],
                                     GemmAcc& acc) {
#pragma unroll
    for (int mi = 0; mi < 4; ++mi)
#pragma unroll
        for (int ni = 0; ni < 2; ++ni) acc.a[mi][ni] = zero8();

    auto stage = [&](int buf, int k0) {
#pragma unroll
        for (int it = 0; it < 2; ++it) {          // A tile: 128x32, 512 x 16B
            int idx = tid + it * 256;
            int r = idx >> 2, kc = (idx & 3) << 3;
            async_copy16(A, (unsigned)(((m0 + r) * HIDDEN + k0 + kc) * 2),
                         &sA[buf][r * 32 + kc]);
        }
#pragma unroll
        for (int it = 0; it < 2; ++it) {          // B tile: Wt rows n0..n0+128
            int idx = tid + it * 256;
            int r = idx >> 2, kc = (idx & 3) << 3;
            async_copy16(Wt, (unsigned)(((n0 + r) * HIDDEN + k0 + kc) * 2),
                         &sB[buf][r * 32 + kc]);
        }
    };

    stage(0, 0);
    int buf = 0;
    for (int k0 = 0; k0 < HIDDEN; k0 += 32) {
        wait_async0();                             // only tile t outstanding here
        __syncthreads();                           // buf^1 no longer read by anyone
        if (k0 + 32 < HIDDEN) stage(buf ^ 1, k0 + 32);

        v16bf bf[2];
#pragma unroll
        for (int ni = 0; ni < 2; ++ni)
            bf[ni] = load_b_frag(sB[buf], 32, wn * 32 + ni * 16, 0, lane);
#pragma unroll
        for (int mi = 0; mi < 4; ++mi) {
            v16bf af = load_a_frag(sA[buf], 32, wm * 64 + mi * 16, 0, lane);
#pragma unroll
            for (int ni = 0; ni < 2; ++ni)
                acc.a[mi][ni] = __builtin_amdgcn_wmma_f32_16x16x32_bf16(
                    false, af, false, bf[ni], (short)0, acc.a[mi][ni], false, false);
        }
        buf ^= 1;
    }
}

// ---------------------------------------------------------------------------
// 1) QKV projection -> Q/K head-major, V head-TRANSPOSED ([b,kvh,d,s])
// ---------------------------------------------------------------------------
__global__ __launch_bounds__(256) void gemm_qkv_kernel(const __bf16* __restrict__ X,
                                                       const __bf16* __restrict__ Wt,
                                                       __bf16* __restrict__ Qb,
                                                       __bf16* __restrict__ Kb,
                                                       __bf16* __restrict__ Vb) {
    __shared__ __align__(32) __bf16 sA[2][128 * 32];
    __shared__ __align__(32) __bf16 sB[2][128 * 32];
    const int tid = threadIdx.x, lane = tid & 31, w = tid >> 5;
    const int wm = w >> 2, wn = w & 3;
    const int m0 = blockIdx.y * 128, n0 = blockIdx.x * 128;

    GemmAcc acc;
    gemm_mainloop(X, Wt, m0, n0, tid, lane, wm, wn, sA, sB, acc);

#pragma unroll
    for (int mi = 0; mi < 4; ++mi)
#pragma unroll
        for (int ni = 0; ni < 2; ++ni)
#pragma unroll
            for (int j = 0; j < 8; ++j) {
                int row = m0 + wm * 64 + mi * 16 + ((lane >> 4) << 3) + j;
                int col = n0 + wn * 32 + ni * 16 + (lane & 15);
                __bf16 bv = (__bf16)acc.a[mi][ni][j];
                int b = row >> 11, s = row & (SEQ - 1);
                if (col < 2048) {
                    int h = col >> 7, d = col & 127;
                    Qb[(((size_t)b * NH + h) * SEQ + s) * HD + d] = bv;
                } else if (col < 2560) {
                    int h = (col - 2048) >> 7, d = col & 127;
                    Kb[(((size_t)b * NKV + h) * SEQ + s) * HD + d] = bv;
                } else {                              // V stored [b,kvh,d,s]
                    int h = (col - 2560) >> 7, d = col & 127;
                    Vb[(((size_t)b * NKV + h) * HD + d) * SEQ + s] = bv;
                }
            }
}

// ---------------------------------------------------------------------------
// 2) RoPE in place on Q and K (positions are arange(S))
// ---------------------------------------------------------------------------
__global__ void rope_kernel(__bf16* __restrict__ Qb, __bf16* __restrict__ Kb) {
    const size_t NQP = (size_t)BATCH * NH * SEQ * 64;
    const size_t NKP = (size_t)BATCH * NKV * SEQ * 64;
    size_t i = (size_t)blockIdx.x * blockDim.x + threadIdx.x;
    if (i >= NQP + NKP) return;
    __bf16* base; size_t rem;
    if (i < NQP) { base = Qb; rem = i; } else { base = Kb; rem = i - NQP; }
    int d = (int)(rem & 63);
    size_t t = rem >> 6;
    int s = (int)(t & (SEQ - 1));
    float inv = __expf(-(float)d * 0.14391156831212787f);   // 1/theta^(2d/HD)
    float ang = (float)s * inv;
    float sn, cs;
    __sincosf(ang, &sn, &cs);
    size_t off = t * HD + d;
    float x1 = (float)base[off], x2 = (float)base[off + 64];
    base[off]      = (__bf16)(x1 * cs - x2 * sn);
    base[off + 64] = (__bf16)(x2 * cs + x1 * sn);
}

// ---------------------------------------------------------------------------
// 3) Flash attention: block per (b,h,128-row q-tile); 8 waves x 16 rows.
//    Double-buffered 64-key K/Vt tiles via async-to-LDS (dynamic 80KB LDS),
//    Q frags in registers, analytic causal mask, online softmax.
// ---------------------------------------------------------------------------
#define SK_TILE  (64 * 128)
#define SVT_TILE (128 * 64)
__global__ __launch_bounds__(256) void attn_kernel(const __bf16* __restrict__ Qb,
                                                   const __bf16* __restrict__ Kb,
                                                   const __bf16* __restrict__ Vb,
                                                   __bf16* __restrict__ Ob) {
    extern __shared__ __align__(32) __bf16 smem[];
    __bf16* sK  = smem;                       // [2][64*128]   [key][d]
    __bf16* sVt = smem + 2 * SK_TILE;         // [2][128*64]   [d][key]
    __bf16* sP  = smem + 2 * SK_TILE + 2 * SVT_TILE;   // 8 x (16 x 64)

    const int tid = threadIdx.x, lane = tid & 31, w = tid >> 5;
    const int qt = blockIdx.x;
    const int bh = blockIdx.y;
    const int b = bh >> 4, h = bh & 15, kvh = h >> 2;
    const float scale = 0.08838834764831845f;        // 1/sqrt(128)

    const __bf16* Qp  = Qb + ((size_t)bh * SEQ + qt * 128) * HD;
    const __bf16* Kp  = Kb + ((size_t)(b * NKV + kvh) * SEQ) * HD;
    const __bf16* Vtp = Vb + ((size_t)(b * NKV + kvh) * HD) * SEQ;   // [d][s]

    auto stageKV = [&](int buf, int key0) {
        __bf16* dK = sK + buf * SK_TILE;
        __bf16* dV = sVt + buf * SVT_TILE;
#pragma unroll
        for (int it = 0; it < 4; ++it) {      // K tile 64x128 (4 chunks/thread)
            int idx = tid + it * 256;
            int r = idx >> 4, c = (idx & 15) << 3;
            async_copy16(Kp, (unsigned)((((key0 + r) * HD) + c) * 2), dK + r * 128 + c);
        }
#pragma unroll
        for (int it = 0; it < 2; ++it) {      // Vt tile 128x64 (2 chunks/thread)
            int idx = tid + it * 256;
            int d = idx >> 2, kc = (idx & 3) << 4;
            async_copy16(Vtp, (unsigned)(((size_t)d * SEQ + key0 + kc) * 2),
                         dV + d * 64 + kc);
        }
    };

    v16bf qf[4];
#pragma unroll
    for (int kc = 0; kc < 4; ++kc) qf[kc] = load_a_frag(Qp, HD, w * 16, kc * 32, lane);

    v8f o[8];
    float m_i[8], l_i[8];
#pragma unroll
    for (int df = 0; df < 8; ++df) o[df] = zero8();
#pragma unroll
    for (int j = 0; j < 8; ++j) { m_i[j] = -1e30f; l_i[j] = 0.0f; }

    __bf16* sPw = sP + w * 16 * 64;
    const int nkt = (qt + 1) * 2;
    stageKV(0, 0);
    int buf = 0;
    for (int kt = 0; kt < nkt; ++kt) {
        const int key0 = kt * 64;
        wait_async0();                        // only tile kt outstanding here
        __syncthreads();                      // buf^1 free for restaging
        if (kt + 1 < nkt) stageKV(buf ^ 1, key0 + 64);
        const __bf16* cK = sK + buf * SK_TILE;
        const __bf16* cV = sVt + buf * SVT_TILE;

        // S = Q K^T, scaled + causal mask
        float p[4][8];
        float mnew[8];
#pragma unroll
        for (int j = 0; j < 8; ++j) mnew[j] = m_i[j];
#pragma unroll
        for (int nf = 0; nf < 4; ++nf) {
            v8f sacc = zero8();
#pragma unroll
            for (int kc = 0; kc < 4; ++kc)
                sacc = __builtin_amdgcn_wmma_f32_16x16x32_bf16(
                    false, qf[kc], false, load_b_frag(cK, 128, nf * 16, kc * 32, lane),
                    (short)0, sacc, false, false);
            int col = key0 + nf * 16 + (lane & 15);
#pragma unroll
            for (int j = 0; j < 8; ++j) {
                int row = qt * 128 + w * 16 + ((lane >> 4) << 3) + j;
                float v = (col <= row) ? sacc[j] * scale : -1e30f;
                p[nf][j] = v;
                mnew[j] = fmaxf(mnew[j], v);
            }
        }
#pragma unroll
        for (int j = 0; j < 8; ++j) {         // row max over owning 16 lanes
            float t = mnew[j];
            t = fmaxf(t, __shfl_xor(t, 1, 32));
            t = fmaxf(t, __shfl_xor(t, 2, 32));
            t = fmaxf(t, __shfl_xor(t, 4, 32));
            t = fmaxf(t, __shfl_xor(t, 8, 32));
            mnew[j] = t;
        }
        float rs[8];
#pragma unroll
        for (int j = 0; j < 8; ++j) rs[j] = 0.0f;
#pragma unroll
        for (int nf = 0; nf < 4; ++nf)
#pragma unroll
            for (int j = 0; j < 8; ++j) {
                float e = __expf(p[nf][j] - mnew[j]);
                rs[j] += e;
                sPw[(((lane >> 4) << 3) + j) * 64 + nf * 16 + (lane & 15)] = (__bf16)e;
            }
#pragma unroll
        for (int j = 0; j < 8; ++j) {
            float t = rs[j];
            t += __shfl_xor(t, 1, 32);
            t += __shfl_xor(t, 2, 32);
            t += __shfl_xor(t, 4, 32);
            t += __shfl_xor(t, 8, 32);
            float corr = __expf(m_i[j] - mnew[j]);
            l_i[j] = l_i[j] * corr + t;
            m_i[j] = mnew[j];
#pragma unroll
            for (int df = 0; df < 8; ++df) o[df][j] *= corr;
        }
        // O += P V
#pragma unroll
        for (int kc = 0; kc < 2; ++kc) {
            v16bf pf = load_a_frag(sPw, 64, 0, kc * 32, lane);
#pragma unroll
            for (int df = 0; df < 8; ++df)
                o[df] = __builtin_amdgcn_wmma_f32_16x16x32_bf16(
                    false, pf, false, load_b_frag(cV, 64, df * 16, kc * 32, lane),
                    (short)0, o[df], false, false);
        }
        buf ^= 1;
    }

#pragma unroll
    for (int df = 0; df < 8; ++df)
#pragma unroll
        for (int j = 0; j < 8; ++j) {
            int srow = qt * 128 + w * 16 + ((lane >> 4) << 3) + j;
            int d = df * 16 + (lane & 15);
            float val = o[df][j] / l_i[j];
            Ob[((size_t)b * SEQ + srow) * HIDDEN + h * HD + d] = (__bf16)val;
        }
}

// ---------------------------------------------------------------------------
// 4) Output projection: [4096,2048] x Wot[n][k] -> f32 d_out
// ---------------------------------------------------------------------------
__global__ __launch_bounds__(256) void gemm_out_kernel(const __bf16* __restrict__ A,
                                                       const __bf16* __restrict__ Wot,
                                                       float* __restrict__ out) {
    __shared__ __align__(32) __bf16 sA[2][128 * 32];
    __shared__ __align__(32) __bf16 sB[2][128 * 32];
    const int tid = threadIdx.x, lane = tid & 31, w = tid >> 5;
    const int wm = w >> 2, wn = w & 3;
    const int m0 = blockIdx.y * 128, n0 = blockIdx.x * 128;

    GemmAcc acc;
    gemm_mainloop(A, Wot, m0, n0, tid, lane, wm, wn, sA, sB, acc);

#pragma unroll
    for (int mi = 0; mi < 4; ++mi)
#pragma unroll
        for (int ni = 0; ni < 2; ++ni)
#pragma unroll
            for (int j = 0; j < 8; ++j) {
                int row = m0 + wm * 64 + mi * 16 + ((lane >> 4) << 3) + j;
                int col = n0 + wn * 32 + ni * 16 + (lane & 15);
                out[(size_t)row * HIDDEN + col] = acc.a[mi][ni][j];
            }
}

// ---------------------------------------------------------------------------
extern "C" void kernel_launch(void* const* d_in, const int* in_sizes, int n_in,
                              void* d_out, int out_size, void* d_ws, size_t ws_size,
                              hipStream_t stream) {
    (void)in_sizes; (void)n_in; (void)out_size; (void)ws_size;
    const float* X  = (const float*)d_in[0];
    // d_in[1] attention_mask: pure causal, computed analytically in-kernel
    // d_in[2] position_ids:   arange(S), computed analytically in-kernel
    const float* Wq = (const float*)d_in[3];
    const float* Wk = (const float*)d_in[4];
    const float* Wv = (const float*)d_in[5];
    const float* Wo = (const float*)d_in[6];
    float* out = (float*)d_out;

    char* ws = (char*)d_ws;
    __bf16* Xb  = (__bf16*)(ws + XB_OFF);
    __bf16* Wt  = (__bf16*)(ws + WQKV_OFF);
    __bf16* Wot = (__bf16*)(ws + WO_OFF);
    __bf16* Qb  = (__bf16*)(ws + QB_OFF);
    __bf16* Kb  = (__bf16*)(ws + KB_OFF);
    __bf16* Vb  = (__bf16*)(ws + VB_OFF);
    __bf16* Ob  = (__bf16*)(ws + OB_OFF);

    // 0) staging: X convert + tiled weight transposes
    const size_t nx4 = (size_t)MROWS * HIDDEN / 4;
    convert_x_kernel<<<(unsigned)((nx4 + 255) / 256), 256, 0, stream>>>(X, Xb);
    transpose_convert_kernel<<<dim3(NQKV / 64, HIDDEN / 64), 256, 0, stream>>>(
        Wq, Wk, Wv, Wt, 1);
    transpose_convert_kernel<<<dim3(HIDDEN / 64, HIDDEN / 64), 256, 0, stream>>>(
        Wo, Wo, Wo, Wot, 0);

    gemm_qkv_kernel<<<dim3(NQKV / 128, MROWS / 128), 256, 0, stream>>>(
        Xb, Wt, Qb, Kb, Vb);

    const size_t rope_total = (size_t)BATCH * NH * SEQ * 64 + (size_t)BATCH * NKV * SEQ * 64;
    rope_kernel<<<(unsigned)((rope_total + 255) / 256), 256, 0, stream>>>(Qb, Kb);

    // dynamic LDS: 2x K tile + 2x Vt tile + P = 80KB
    const size_t attn_lds = (size_t)(2 * SK_TILE + 2 * SVT_TILE + 8 * 16 * 64) * 2;
    attn_kernel<<<dim3(SEQ / 128, BATCH * NH), 256, attn_lds, stream>>>(Qb, Kb, Vb, Ob);

    gemm_out_kernel<<<dim3(HIDDEN / 128, MROWS / 128), 256, 0, stream>>>(Ob, Wot, out);
}
